// RTNCustomLinear_72997264163015
// MI455X (gfx1250) — matrix operand order repaired and checked by reading, not compiled, not run on hardware
//
#include <hip/hip_runtime.h>

// RTN 4-bit dequant GEMM for MI455X (gfx1250, wave32, WMMA).
// y[m, o] = sum_k x[m,k] * (qw[o,k] * scales[o, k/128]) + bias[o]
// M = 2*2048 = 4096 tokens, K = 4096, N = 11008.
// Compute-bound (AI ~870 flop/B): run on v_wmma_f32_16x16x32_f16,
// dequantizing weights to f16 on the fly and converting x fp32->f16.

#define IN_F   4096
#define OUT_F  11008
#define GROUP  128

#define BM 128
#define BN 128
#define BK 64
#define LDK (BK + 8)   // padded LDS row stride in halves (144 B: 16B-aligned, conflict-free)

typedef __attribute__((ext_vector_type(16))) _Float16 v16h;
typedef __attribute__((ext_vector_type(8)))  _Float16 v8h;
typedef __attribute__((ext_vector_type(8)))  float    v8f;
typedef __attribute__((ext_vector_type(4)))  float    v4f;
typedef __attribute__((ext_vector_type(4)))  int      v4i;

__global__ __launch_bounds__(256) void RTNCustomLinear_72997264163015_kernel(
    const float* __restrict__ x,       // [M, IN_F] fp32
    const int*   __restrict__ qw,      // [OUT_F, IN_F] int32 codes in [-8,7]
    const float* __restrict__ scales,  // [OUT_F, IN_F/GROUP]
    const float* __restrict__ bias,    // [OUT_F]
    float*       __restrict__ out)     // [M, OUT_F]
{
    __shared__ _Float16 Asm[BM * LDK];   // x tile, f16,  [128][72]
    __shared__ _Float16 Bsm[BN * LDK];   // dequant W tile, f16, [128][72]

    const int t    = threadIdx.x;
    const int lane = t & 31;
    const int wave = t >> 5;

    const int blockM = blockIdx.y * BM;
    const int blockN = blockIdx.x * BN;

    // Wave sub-tile: 4 waves along M (32 rows each), 2 along N (64 cols each).
    const int wm = (wave & 3) * 32;
    const int wn = (wave >> 2) * 64;

    // ---- staging coordinates: 2 threads per row, 32 contiguous K each ----
    const int srow  = t >> 1;            // 0..127
    const int skseg = (t & 1) * 32;      // 0 or 32

    const float* xg = x  + (size_t)(blockM + srow) * IN_F + skseg;
    const int*   wg = qw + (size_t)(blockN + srow) * IN_F + skseg;
    const float* sg = scales + (size_t)(blockN + srow) * (IN_F / GROUP);

    _Float16* As = &Asm[srow * LDK + skseg];
    _Float16* Bs = &Bsm[srow * LDK + skseg];

    // ---- fragment coordinates (ISA 7.12.2 wave32 layouts) ----
    const int frow  = lane & 15;          // A: M row / B: N col within 16-tile
    const int akoff = (lane >> 4) * 8;    // A K-half select
    const int bkoff = (lane >> 4) * 8;    // B K-half select

    v8f c[2][4] = {};                     // 2 M-tiles x 4 N-tiles, fp32 accum

    for (int k0 = 0; k0 < IN_F; k0 += BK) {
        // -------- global loads into registers --------
        float xv[32];
        int   qv[32];
#pragma unroll
        for (int i = 0; i < 8; ++i)
            *(v4f*)(xv + 4 * i) = *(const v4f*)(xg + k0 + 4 * i);
#pragma unroll
        for (int i = 0; i < 8; ++i)
            *(v4i*)(qv + 4 * i) = *(const v4i*)(wg + k0 + 4 * i);
        const float sc = sg[(k0 + skseg) >> 7];   // one group per 32-wide segment

        if (k0 + BK < IN_F) {                      // gfx1250 global_prefetch_b8
            __builtin_prefetch(xg + k0 + BK, 0, 3);
            __builtin_prefetch(wg + k0 + BK, 0, 3);
        }

        __syncthreads();   // previous iteration's LDS readers done

        // -------- convert + store to LDS (16-byte chunks) --------
#pragma unroll
        for (int cch = 0; cch < 4; ++cch) {
            v8h ha, hb;
#pragma unroll
            for (int j = 0; j < 8; ++j) {
                ha[j] = (_Float16)xv[cch * 8 + j];
                hb[j] = (_Float16)((float)qv[cch * 8 + j] * sc);
            }
            *(v8h*)(As + cch * 8) = ha;
            *(v8h*)(Bs + cch * 8) = hb;
        }

        __syncthreads();   // tile visible to all waves

        // -------- 2 WMMA K-steps over the 64-wide LDS tile --------
#pragma unroll
        for (int ks = 0; ks < BK; ks += 32) {
            v16h af[2], bf[4];
#pragma unroll
            for (int mi = 0; mi < 2; ++mi) {
                const _Float16* ap = &Asm[(wm + mi * 16 + frow) * LDK + ks + akoff];
                *((v8h*)&af[mi])     = *(const v8h*)(ap);       // K = kb..kb+7
                *((v8h*)&af[mi] + 1) = *(const v8h*)(ap + 16);  // K = 16+kb..
            }
#pragma unroll
            for (int ni = 0; ni < 4; ++ni) {
                const _Float16* bp = &Bsm[(wn + ni * 16 + frow) * LDK + ks + bkoff];
                *((v8h*)&bf[ni])     = *(const v8h*)(bp);
                *((v8h*)&bf[ni] + 1) = *(const v8h*)(bp + 16);
            }
#pragma unroll
            for (int mi = 0; mi < 2; ++mi)
#pragma unroll
                for (int ni = 0; ni < 4; ++ni)
                    c[mi][ni] = __builtin_amdgcn_wmma_f32_16x16x32_f16(
                        false, af[mi], false, bf[ni],
                        (short)0, c[mi][ni], false, false);
        }
    }

    // -------- epilogue: bias add + store (C/D layout: VGPR r -> M = (lane>>4)*8 + r) --------
    const int ml = (lane >> 4) * 8;
    const int nl = lane & 15;
#pragma unroll
    for (int ni = 0; ni < 4; ++ni) {
        const int ng = blockN + wn + ni * 16 + nl;
        const float bv = bias[ng];
#pragma unroll
        for (int mi = 0; mi < 2; ++mi) {
            const int mg = blockM + wm + mi * 16 + ml;
            float* op = out + (size_t)mg * OUT_F + ng;
#pragma unroll
            for (int r = 0; r < 8; ++r)
                op[(size_t)r * OUT_F] = c[mi][ni][r] + bv;
        }
    }
}

extern "C" void kernel_launch(void* const* d_in, const int* in_sizes, int n_in,
                              void* d_out, int out_size, void* d_ws, size_t ws_size,
                              hipStream_t stream) {
    (void)n_in; (void)out_size; (void)d_ws; (void)ws_size;
    const float* x      = (const float*)d_in[0];
    const int*   qw     = (const int*)  d_in[1];
    const float* scales = (const float*)d_in[2];
    const float* bias   = (const float*)d_in[3];
    float*       out    = (float*)d_out;

    const int M = in_sizes[0] / IN_F;    // 2*2048 = 4096
    dim3 grid(OUT_F / BN, M / BM);       // (86, 32)
    RTNCustomLinear_72997264163015_kernel<<<grid, 256, 0, stream>>>(
        x, qw, scales, bias, out);
}
